// WindAttention4D_71451075937022
// MI455X (gfx1250) — matrix operand order repaired and checked by reading, not compiled, not run on hardware
//
#include <hip/hip_runtime.h>
#include <hip/hip_bf16.h>

typedef __attribute__((ext_vector_type(16))) __bf16 v16bf;
typedef __attribute__((ext_vector_type(8)))  __bf16 v8bf;
typedef __attribute__((ext_vector_type(4)))  __bf16 v4bf;
typedef __attribute__((ext_vector_type(8)))  float  v8f;
typedef __attribute__((ext_vector_type(4)))  float  f32x4;

#define DIM    256
#define HEADS  8
#define DH     32
#define NTOK   144
#define NWIN   64
#define BATCH  1024
#define TC     768   // 3*DIM

#define QK_LD  520   // Q|K region: [144][520] bf16 (1040B rows, 260%64=4 bank skew)
#define VT_LD  168   // V^T region: [256][168] bf16 (336B rows, 84%64=20 -> 16 lanes distinct banks)
#define XS_LD  264   // x staging / attn-out LD (528B rows)
#define ATT_LD 264
#define PR_LD  144   // probs LD (288B rows)

#define QK_ELEMS  (NTOK * QK_LD)    // 74880
#define VT_ELEMS  (DIM * VT_LD)     // 43008
#define AUX_ELEMS (NTOK * XS_LD)    // 38016 (>= probs 8*16*144 = 18432)
#define SMEM_BYTES ((QK_ELEMS + VT_ELEMS + AUX_ELEMS) * 2)  // 311808 B

#define QSCALE 0.17677669529663687f  // Dh^-0.5, Dh=32

#define WMMA_BF16(a, b, c) \
  __builtin_amdgcn_wmma_f32_16x16x32_bf16(false, (a), false, (b), (short)0, (c), false, false)

// 16-bit A-matrix (16x32): per-lane one contiguous 16-elem run, elem e -> K = hf*16 + e.
__device__ __forceinline__ v16bf load_a16(const __bf16* s, int row, int k0, int ld, int hf) {
  const __bf16* p = s + row * ld + k0 + hf * 16;
  v8bf lo = *(const v8bf*)(p);
  v8bf hi = *(const v8bf*)(p + 8);
  return __builtin_shufflevector(lo, hi, 0,1,2,3,4,5,6,7,8,9,10,11,12,13,14,15);
}

// 16-bit B-matrix (32x16): lane = column, elem e -> K = hf*16 + e.
// With K-major (transposed) storage this is one contiguous 32B run per lane.
__device__ __forceinline__ v16bf load_b16(const __bf16* p) {
  v8bf lo = *(const v8bf*)(p);
  v8bf hi = *(const v8bf*)(p + 8);
  return __builtin_shufflevector(lo, hi, 0,1,2,3,4,5,6,7,8,9,10,11,12,13,14,15);
}

__device__ __forceinline__ v8f cinit(float bval) {
  v8f c;
#pragma unroll
  for (int v = 0; v < 8; ++v) c[v] = bval;
  return c;
}

// ---- pre-pass: transpose + convert weights to bf16; fold Q-scale into wq / bq ----
__global__ void __launch_bounds__(256)
cvt_weights(const float* __restrict__ w_qkv, const float* __restrict__ b_qkv,
            const float* __restrict__ w_proj,
            __bf16* __restrict__ wqkvT, __bf16* __restrict__ wprojT,
            float* __restrict__ bias_s) {
  int idx = blockIdx.x * 256 + threadIdx.x;   // grid covers TC*DIM
  if (idx < TC * DIM) {
    int j = idx >> 8;        // original column (0..767)
    int k = idx & (DIM - 1); // original row    (0..255)
    float w = w_qkv[k * TC + j];
    if (j < DIM) w *= QSCALE;                  // pre-scale Q columns
    wqkvT[idx] = (__bf16)w;                    // wqkvT[j][k]
  }
  if (idx < DIM * DIM) {
    int j = idx >> 8;
    int k = idx & (DIM - 1);
    wprojT[idx] = (__bf16)w_proj[k * DIM + j]; // wprojT[j][k]
  }
  if (idx < TC) {
    float bv = b_qkv[idx];
    if (idx < DIM) bv *= QSCALE;               // pre-scale Q bias
    bias_s[idx] = bv;
  }
}

__global__ void __launch_bounds__(256)
wind_attn_fused(const float* __restrict__ x, const float* __restrict__ mask,
                const __bf16* __restrict__ wqkvT, const float* __restrict__ bias_s,
                const __bf16* __restrict__ wprojT, const float* __restrict__ b_proj,
                float* __restrict__ out) {
  extern __shared__ char smem_raw[];
  __bf16* qk   = (__bf16*)smem_raw;          // [144][QK_LD]: Q cols 0..255, K cols 256..511
  __bf16* vT   = qk + QK_ELEMS;              // [256][VT_LD]: V transposed [d][token]
  __bf16* aux  = vT + VT_ELEMS;              // phase1: x bf16 [144][XS_LD]; phase2: probs
  __bf16* attn = qk;                         // phase3: reuse QK region as [144][ATT_LD]

  const int b    = blockIdx.x;
  const int tid  = threadIdx.x;
  const int wave = tid >> 5;
  const int lane = tid & 31;
  const int l16  = lane & 15;
  const int hf   = (lane >> 4) & 1;

  // ---------------- Phase 0: stage x[b] -> aux bf16; zero V^T token-pad cols ----------------
  {
    const float* xb = x + (size_t)b * (NTOK * DIM);
    for (int idx = tid; idx < NTOK * DIM / 4; idx += 256) {
      int f   = idx * 4;
      int row = f >> 8;
      int col = f & (DIM - 1);
      f32x4 v = *(const f32x4*)(xb + f);
      v4bf d; d[0] = (__bf16)v.x; d[1] = (__bf16)v.y; d[2] = (__bf16)v.z; d[3] = (__bf16)v.w;
      *(v4bf*)(aux + row * XS_LD + col) = d;
    }
    v8bf z = {};
    *(v8bf*)(vT + tid * VT_LD + 144) = z;
    *(v8bf*)(vT + tid * VT_LD + 152) = z;
  }
  __syncthreads();

  // ---------------- Phase 1: qkv = x @ wqkv + b (Q pre-scaled in pre-pass) ----------------
  // Q|K tiles: straight-line epilogue into qk region
  for (int nt = wave; nt < 2 * DIM / 16; nt += 8) {   // 4 per wave
    const int col = nt * 16 + l16;
    v16bf Breg[8];
#pragma unroll
    for (int ks = 0; ks < 8; ++ks)
      Breg[ks] = load_b16(wqkvT + col * DIM + ks * 32 + hf * 16);
    const float bias = bias_s[col];
    for (int mt = 0; mt < 9; ++mt) {
      v8f acc = cinit(bias);                          // bias via C operand
#pragma unroll
      for (int ks = 0; ks < 8; ++ks) {
        v16bf a = load_a16(aux, mt * 16 + l16, ks * 32, XS_LD, hf);
        acc = WMMA_BF16(a, Breg[ks], acc);
      }
#pragma unroll
      for (int v = 0; v < 8; ++v)
        qk[(mt * 16 + hf * 8 + v) * QK_LD + col] = (__bf16)acc[v];
    }
  }
  // V tiles: stored transposed into vT region
  for (int nt = 2 * DIM / 16 + wave; nt < TC / 16; nt += 8) {  // 2 per wave
    const int col = nt * 16 + l16;
    v16bf Breg[8];
#pragma unroll
    for (int ks = 0; ks < 8; ++ks)
      Breg[ks] = load_b16(wqkvT + col * DIM + ks * 32 + hf * 16);
    const float bias = bias_s[col];
    const int vd = col - 2 * DIM;
    for (int mt = 0; mt < 9; ++mt) {
      v8f acc = cinit(bias);
#pragma unroll
      for (int ks = 0; ks < 8; ++ks) {
        v16bf a = load_a16(aux, mt * 16 + l16, ks * 32, XS_LD, hf);
        acc = WMMA_BF16(a, Breg[ks], acc);
      }
#pragma unroll
      for (int v = 0; v < 8; ++v)
        vT[vd * VT_LD + mt * 16 + hf * 8 + v] = (__bf16)acc[v];
    }
  }
  __syncthreads();

  // ---------------- Phase 2: per (head, q-tile) attention ----------------
  __bf16* probs = aux + wave * (16 * PR_LD);                 // per-wave [16][144]
  const float* maskw = mask + (size_t)(b & (NWIN - 1)) * (NTOK * NTOK);
  v16bf oacc[9];                                             // packed bf16 outputs

#pragma unroll
  for (int i = 0; i < 9; ++i) {
    const int t  = wave + i * 8;
    const int h  = t / 9;
    const int qt = t - h * 9;

    v16bf aq = load_a16(qk, qt * 16 + l16, h * DH, QK_LD, hf);

    // scores: mask injected via the WMMA C operand
    v8f sc[9];
#pragma unroll
    for (int j = 0; j < 9; ++j) {
      v16bf bk = load_b16(qk + (j * 16 + l16) * QK_LD + DIM + h * DH + hf * 16);
      v8f s;
#pragma unroll
      for (int v = 0; v < 8; ++v)
        s[v] = maskw[(qt * 16 + hf * 8 + v) * NTOK + j * 16 + l16];
      sc[j] = WMMA_BF16(aq, bk, s);
    }

    // softmax over 144 cols (row lives in 16 lanes of one half + 9 tiles)
#pragma unroll
    for (int v = 0; v < 8; ++v) {
      float m = -3.0e38f;
#pragma unroll
      for (int j = 0; j < 9; ++j) m = fmaxf(m, sc[j][v]);
#pragma unroll
      for (int off = 8; off >= 1; off >>= 1) m = fmaxf(m, __shfl_xor(m, off, 32));
      float ssum = 0.0f;
#pragma unroll
      for (int j = 0; j < 9; ++j) { float p = __expf(sc[j][v] - m); sc[j][v] = p; ssum += p; }
#pragma unroll
      for (int off = 8; off >= 1; off >>= 1) ssum += __shfl_xor(ssum, off, 32);
      const float r = 1.0f / ssum;
#pragma unroll
      for (int j = 0; j < 9; ++j)
        probs[(hf * 8 + v) * PR_LD + j * 16 + l16] = (__bf16)(sc[j][v] * r);
    }

    // PV: out(16x32) = P(16x144) @ V(144x32); 4 full K32 chunks + 1 zero-padded K16 chunk
    v8f o0 = {}, o1 = {};
#pragma unroll
    for (int kc = 0; kc < 5; ++kc) {
      v16bf ap;
      {
        const __bf16* p = probs + l16 * PR_LD + kc * 32 + hf * 16;
        v8bf lo = {};
        v8bf hi = {};
        if (kc < 4 || hf == 0) { lo = *(const v8bf*)(p); hi = *(const v8bf*)(p + 8); }
        ap = __builtin_shufflevector(lo, hi, 0,1,2,3,4,5,6,7,8,9,10,11,12,13,14,15);
      }
#pragma unroll
      for (int dt = 0; dt < 2; ++dt) {
        v16bf bv = load_b16(vT + (h * DH + dt * 16 + l16) * VT_LD + kc * 32 + hf * 16);
        if (dt == 0) o0 = WMMA_BF16(ap, bv, o0);
        else         o1 = WMMA_BF16(ap, bv, o1);
      }
    }
    v16bf ob;
#pragma unroll
    for (int v = 0; v < 8; ++v) { ob[v] = (__bf16)o0[v]; ob[8 + v] = (__bf16)o1[v]; }
    oacc[i] = ob;
  }
  __syncthreads();   // all waves done reading Q/K/V -> safe to overwrite

  // dump staged attention outputs over the QK region as [144][ATT_LD] bf16
#pragma unroll
  for (int i = 0; i < 9; ++i) {
    const int t  = wave + i * 8;
    const int h  = t / 9;
    const int qt = t - h * 9;
#pragma unroll
    for (int dt = 0; dt < 2; ++dt)
#pragma unroll
      for (int v = 0; v < 8; ++v)
        attn[(qt * 16 + hf * 8 + v) * ATT_LD + h * DH + dt * 16 + l16] = oacc[i][dt * 8 + v];
  }
  __syncthreads();

  // ---------------- Phase 3: out = attn @ w_proj + b_proj (f32 out) ----------------
  {
    float* outb = out + (size_t)b * (NTOK * DIM);
#pragma unroll
    for (int nn = 0; nn < 2; ++nn) {
      const int nt  = wave + nn * 8;     // 16 N-tiles over 8 waves
      const int col = nt * 16 + l16;
      v16bf Breg[8];
#pragma unroll
      for (int ks = 0; ks < 8; ++ks)
        Breg[ks] = load_b16(wprojT + col * DIM + ks * 32 + hf * 16);
      const float bias = b_proj[col];
      for (int mt = 0; mt < 9; ++mt) {
        v8f acc = cinit(bias);                       // bias via C operand
#pragma unroll
        for (int ks = 0; ks < 8; ++ks) {
          v16bf a = load_a16(attn, mt * 16 + l16, ks * 32, ATT_LD, hf);
          acc = WMMA_BF16(a, Breg[ks], acc);
        }
#pragma unroll
        for (int v = 0; v < 8; ++v)
          outb[(mt * 16 + hf * 8 + v) * DIM + col] = acc[v];
      }
    }
  }
}

extern "C" void kernel_launch(void* const* d_in, const int* in_sizes, int n_in,
                              void* d_out, int out_size, void* d_ws, size_t ws_size,
                              hipStream_t stream) {
  (void)in_sizes; (void)n_in; (void)out_size; (void)ws_size;
  const float* x      = (const float*)d_in[0];
  const float* mask   = (const float*)d_in[1];
  const float* w_qkv  = (const float*)d_in[2];
  const float* b_qkv  = (const float*)d_in[3];
  const float* w_proj = (const float*)d_in[4];
  const float* b_proj = (const float*)d_in[5];
  float* out = (float*)d_out;

  __bf16* wqkvT  = (__bf16*)d_ws;                 // [768][256] bf16 = 384 KB
  __bf16* wprojT = wqkvT + (size_t)TC * DIM;      // [256][256] bf16 = 128 KB
  float*  bias_s = (float*)(wprojT + (size_t)DIM * DIM);  // [768] f32

  hipLaunchKernelGGL(cvt_weights, dim3((TC * DIM) / 256), dim3(256), 0, stream,
                     w_qkv, b_qkv, w_proj, wqkvT, wprojT, bias_s);
  hipLaunchKernelGGL(wind_attn_fused, dim3(BATCH), dim3(256), SMEM_BYTES, stream,
                     x, mask, wqkvT, bias_s, wprojT, b_proj, out);
}
